// MambaLayer_46815143527007
// MI455X (gfx1250) — matrix operand order repaired and checked by reading, not compiled, not run on hardware
//
#include <hip/hip_runtime.h>
#include <stdint.h>

typedef __attribute__((ext_vector_type(16))) __bf16          v16bf;
typedef __attribute__((ext_vector_type(8)))  __bf16          v8bf;
typedef __attribute__((ext_vector_type(8)))  float           v8f;
typedef __attribute__((ext_vector_type(16))) unsigned short  v16us;

#define DEV __device__ __forceinline__

// ---------------- scalar helpers ----------------
DEV unsigned short f2bf(float f) {
    unsigned int u = __float_as_uint(f);
    unsigned int r = u + 0x7FFFu + ((u >> 16) & 1u);   // RNE
    return (unsigned short)(r >> 16);
}
DEV float bf2f(unsigned short h) { return __uint_as_float(((unsigned int)h) << 16); }
DEV float sigmoidf_(float x) { return 1.0f / (1.0f + __expf(-x)); }
DEV float siluf_(float x)    { return x * sigmoidf_(x); }

// ---------------- problem constants ----------------
constexpr int B_ = 8, C_ = 512, L_ = 256;       // H=W=16
constexpr int DIM = 128, CHS = 1024, K8 = 8;
constexpr int DIN = 2048, NIN = 4096, XDB = 96; // dt_rank=64, d_state=16
constexpr float BN_SCALE = 0.9999950000374997f; // 1/sqrt(1+1e-5)

// ---------------- workspace layout ----------------
constexpr size_t OFF_XC   = 0;                              // f32 (B,CHS,L)
constexpr size_t SZ_XC    = (size_t)B_*CHS*L_*4;
constexpr size_t OFF_GW   = OFF_XC + SZ_XC;                 // f32 (B,8) padded
constexpr size_t OFF_POOL = OFF_GW + 4096;                  // f32 (B,C,275)
constexpr size_t SZ_POOL  = (size_t)B_*C_*275*4;
constexpr size_t OFF_YBR  = OFF_POOL + SZ_POOL;             // f32 (B,DIM,275)
constexpr size_t SZ_YBR   = (size_t)B_*DIM*275*4;
constexpr size_t OFF_XG   = OFF_YBR + SZ_YBR;               // bf16 (B*K8,L,CHS)
constexpr size_t SZ_XG    = (size_t)B_*K8*L_*CHS*2;
constexpr size_t OFF_WIN  = OFF_XG + SZ_XG;                 // bf16 (K8,NIN,CHS)
constexpr size_t SZ_WIN   = (size_t)K8*NIN*CHS*2;
constexpr size_t OFF_WXP  = OFF_WIN + SZ_WIN;               // bf16 (K8,XDB,DIN)
constexpr size_t SZ_WXP   = (size_t)K8*XDB*DIN*2;
constexpr size_t OFF_WDT  = OFF_WXP + SZ_WXP;               // bf16 (K8,DIN,64)
constexpr size_t SZ_WDT   = (size_t)K8*DIN*64*2;
constexpr size_t OFF_WOUT = OFF_WDT + SZ_WDT;               // bf16 (K8,CHS,DIN)
constexpr size_t SZ_WOUT  = (size_t)K8*CHS*DIN*2;
constexpr size_t OFF_XZ   = OFF_WOUT + SZ_WOUT;             // bf16 (B*K8,L,NIN)
constexpr size_t SZ_XZ    = (size_t)B_*K8*L_*NIN*2;
constexpr size_t OFF_U    = OFF_XZ + SZ_XZ;                 // bf16 (B*K8,L,DIN)
constexpr size_t SZ_U     = (size_t)B_*K8*L_*DIN*2;
constexpr size_t OFF_XDBL = OFF_U + SZ_U;                   // f32 (B*K8,L,96)
constexpr size_t SZ_XDBL  = (size_t)B_*K8*L_*XDB*4;
constexpr size_t OFF_DT   = OFF_XDBL + SZ_XDBL;             // bf16 (B*K8,L,DIN)
constexpr size_t SZ_DT    = (size_t)B_*K8*L_*DIN*2;
constexpr size_t OFF_YG   = OFF_DT + SZ_DT;                 // bf16 (B*K8,L,DIN)
constexpr size_t SZ_YG    = SZ_DT;
constexpr size_t OFF_YS   = OFF_YG + SZ_YG;                 // f32 (B*K8,L,CHS)

// ---------------- f32 -> bf16 convert ----------------
__global__ void k_cvt(const float* __restrict__ s, unsigned short* __restrict__ d, int n) {
    for (int i = blockIdx.x * blockDim.x + threadIdx.x; i < n; i += gridDim.x * blockDim.x)
        d[i] = f2bf(s[i]);
}

// ---------------- pooling pyramid ----------------
DEV void pool_decode(int idx, int& s, int& i, int& j, int& pi) {
    if (idx < 25)       { s = 5;  pi = 1; i = idx / 5;        j = idx % 5; }
    else if (idx < 106) { s = 9;  pi = 2; i = (idx-25) / 9;   j = (idx-25) % 9; }
    else                { s = 13; pi = 3; i = (idx-106) / 13; j = (idx-106) % 13; }
}

__global__ void k_pool(const float* __restrict__ x, float* __restrict__ pooled) {
    int total = B_ * C_ * 275;
    for (int t = blockIdx.x * blockDim.x + threadIdx.x; t < total; t += gridDim.x * blockDim.x) {
        int idx = t % 275, bc = t / 275;
        int s, i, j, pi; pool_decode(idx, s, i, j, pi);
        int hs = i*16/s, he = ((i+1)*16 + s-1)/s;
        int wss = j*16/s, we = ((j+1)*16 + s-1)/s;
        const float* xp = x + (size_t)bc * 256;
        float acc = 0.f;
        for (int h = hs; h < he; ++h)
            for (int w = wss; w < we; ++w) acc += xp[h*16 + w];
        pooled[t] = acc / (float)((he-hs)*(we-wss));
    }
}

__global__ void k_cbr(const float* __restrict__ pooled, const float* __restrict__ pw,
                      const float* __restrict__ bn_g, const float* __restrict__ bn_b,
                      float* __restrict__ ybr) {
    int total = B_ * DIM * 275;
    for (int t = blockIdx.x * blockDim.x + threadIdx.x; t < total; t += gridDim.x * blockDim.x) {
        int idx = t % 275, bo = t / 275;
        int o = bo % DIM, b = bo / DIM;
        int s, i, j, pi; pool_decode(idx, s, i, j, pi);
        const float* pp = pooled + (size_t)b * C_ * 275 + idx;
        const float* wp = pw + ((size_t)pi * DIM + o) * C_;
        float acc = 0.f;
        for (int c = 0; c < C_; ++c) acc += pp[(size_t)c * 275] * wp[c];
        float v = acc * bn_g[pi*DIM + o] * BN_SCALE + bn_b[pi*DIM + o];
        ybr[t] = fminf(fmaxf(v, 0.f), 6.f);
    }
}

// pools[0]: full-res 1x1 conv + clip, spatial mean, broadcast into xc[:, 0:128]
__global__ void k_p0(const float* __restrict__ x, const float* __restrict__ pw,
                     const float* __restrict__ bn_g, const float* __restrict__ bn_b,
                     float* __restrict__ xc) {
    int bo = blockIdx.x; int o = bo % DIM, b = bo / DIM;
    int t = threadIdx.x;                       // spatial position
    const float* xb = x + (size_t)b * C_ * 256 + t;
    const float* wp = pw + (size_t)o * C_;
    float acc = 0.f;
    for (int c = 0; c < C_; ++c) acc += xb[(size_t)c * 256] * wp[c];
    float v = acc * bn_g[o] * BN_SCALE + bn_b[o];
    v = fminf(fmaxf(v, 0.f), 6.f);
    __shared__ float red[256];
    red[t] = v; __syncthreads();
    for (int s = 128; s > 0; s >>= 1) { if (t < s) red[t] += red[t + s]; __syncthreads(); }
    xc[((size_t)b * CHS + o) * 256 + t] = red[0] * (1.f/256.f);
}

// channels 128..1023: bilinear upsample of pyramid + passthrough of x
__global__ void k_assemble(const float* __restrict__ x, const float* __restrict__ ybr,
                           float* __restrict__ xc) {
    int total = B_ * 896 * 256;
    for (int t = blockIdx.x * blockDim.x + threadIdx.x; t < total; t += gridDim.x * blockDim.x) {
        int pos = t % 256, bc = t / 256;
        int ch = bc % 896 + 128, b = bc / 896;
        float v;
        if (ch >= 512) {
            v = x[((size_t)b * C_ + (ch - 512)) * 256 + pos];
        } else {
            int si = (ch - 128) / DIM, o = (ch - 128) % DIM;
            int s   = (si == 0) ? 5 : (si == 1) ? 9 : 13;
            int off = (si == 0) ? 0 : (si == 1) ? 25 : 106;
            int h = pos >> 4, w = pos & 15;
            float fh = (h + 0.5f) * s * (1.f/16.f) - 0.5f;
            float fw = (w + 0.5f) * s * (1.f/16.f) - 0.5f;
            int h0 = (int)floorf(fh); float wh = fh - (float)h0; int h1 = h0 + 1;
            int w0 = (int)floorf(fw); float ww = fw - (float)w0; int w1 = w0 + 1;
            h0 = h0 < 0 ? 0 : (h0 > s-1 ? s-1 : h0);
            h1 = h1 < 0 ? 0 : (h1 > s-1 ? s-1 : h1);
            w0 = w0 < 0 ? 0 : (w0 > s-1 ? s-1 : w0);
            w1 = w1 < 0 ? 0 : (w1 > s-1 ? s-1 : w1);
            const float* yb = ybr + ((size_t)b * DIM + o) * 275 + off;
            float v00 = yb[h0*s + w0], v01 = yb[h0*s + w1];
            float v10 = yb[h1*s + w0], v11 = yb[h1*s + w1];
            v = (1.f-wh)*((1.f-ww)*v00 + ww*v01) + wh*((1.f-ww)*v10 + ww*v11);
        }
        xc[((size_t)b * CHS + ch) * 256 + pos] = v;
    }
}

// ---------------- gating: spatial mean -> logits -> softmax -> top4 ----------------
__global__ void k_gate(const float* __restrict__ xc, const float* __restrict__ gate_w,
                       float* __restrict__ gw) {
    int b = blockIdx.x, t = threadIdx.x;
    __shared__ float sm[CHS];
    for (int c = t; c < CHS; c += 256) {
        const float* p = xc + ((size_t)b * CHS + c) * 256;
        float a = 0.f;
        for (int i = 0; i < 256; ++i) a += p[i];
        sm[c] = a * (1.f/256.f);
    }
    __syncthreads();
    __shared__ float sl[8];
    if (t < 8) {
        const float* g = gate_w + (size_t)t * CHS;
        float a = 0.f;
        for (int c = 0; c < CHS; ++c) a += sm[c] * g[c];
        sl[t] = a;
    }
    __syncthreads();
    if (t == 0) {
        float mx = sl[0];
        for (int i = 1; i < 8; ++i) mx = fmaxf(mx, sl[i]);
        float e[8], sum = 0.f;
        for (int i = 0; i < 8; ++i) { e[i] = __expf(sl[i] - mx); sum += e[i]; }
        float sc[8];
        for (int i = 0; i < 8; ++i) sc[i] = e[i] / sum;
        float out[8] = {0,0,0,0,0,0,0,0};
        bool used[8] = {false,false,false,false,false,false,false,false};
        for (int r = 0; r < 4; ++r) {
            int bi = -1; float bv = -1e30f;
            for (int i = 0; i < 8; ++i) if (!used[i] && sc[i] > bv) { bv = sc[i]; bi = i; }
            used[bi] = true; out[bi] = bv;
        }
        for (int i = 0; i < 8; ++i) gw[b*8 + i] = out[i];
    }
}

// ---------------- cross-scan + finite-difference gate -> bf16 xg ----------------
DEV float cross_val(const float* __restrict__ xcb, int k, int l, int c) {
    int ll = (k == 2 || k == 3 || k == 6 || k == 7) ? (255 - l) : l;
    int kk = (k == 2) ? 0 : (k == 3) ? 1 : (k == 6) ? 4 : (k == 7) ? 5 : k;
    int h, w;
    if (kk == 0) { h = ll >> 4; w = ll & 15; }
    else {
        int w0 = ll >> 4; h = ll & 15;
        if (kk == 4)      w = (h + w0) & 15;
        else if (kk == 5) w = (w0 - h) & 15;
        else              w = w0;            // kk == 1
    }
    return xcb[(size_t)c * 256 + h*16 + w];
}

__global__ void k_scanfd(const float* __restrict__ xc, const float* __restrict__ ln_g,
                         const float* __restrict__ ln_b, const float* __restrict__ fd_scale,
                         unsigned short* __restrict__ xg) {
    int l = blockIdx.x, k = blockIdx.y, b = blockIdx.z, t = threadIdx.x;
    const float* xcb = xc + (size_t)b * CHS * 256;
    float xv[4], dv[4];
    float s1 = 0.f, s2 = 0.f;
#pragma unroll
    for (int q = 0; q < 4; ++q) {
        int c = t*4 + q;
        float v  = cross_val(xcb, k, l, c);
        float vp = (l > 0) ? cross_val(xcb, k, l-1, c) : 0.f;
        xv[q] = v; dv[q] = v - vp;
        s1 += dv[q]; s2 += dv[q]*dv[q];
    }
    __shared__ float r1[256], r2[256];
    r1[t] = s1; r2[t] = s2; __syncthreads();
    for (int s = 128; s > 0; s >>= 1) {
        if (t < s) { r1[t] += r1[t+s]; r2[t] += r2[t+s]; }
        __syncthreads();
    }
    float mu   = r1[0] * (1.f/1024.f);
    float var  = r2[0] * (1.f/1024.f) - mu*mu;
    float rstd = rsqrtf(var + 1e-5f);
    __syncthreads();
    float sn = 0.f;
    float dnv[4];
#pragma unroll
    for (int q = 0; q < 4; ++q) {
        int c = t*4 + q;
        dnv[q] = (dv[q] - mu) * rstd * ln_g[c] + ln_b[c];
        sn += dnv[q]*dnv[q];
    }
    r1[t] = sn; __syncthreads();
    for (int s = 128; s > 0; s >>= 1) { if (t < s) r1[t] += r1[t+s]; __syncthreads(); }
    float nd   = sqrtf(r1[0]) * fd_scale[0];
    float gate = 0.2f + 0.8f * tanhf(fabsf(nd));
    unsigned short* o = xg + (((size_t)(b*8 + k) * 256 + l) * CHS) + t*4;
#pragma unroll
    for (int q = 0; q < 4; ++q) o[q] = f2bf(xv[q] * gate);
}

// ---------------- WMMA fragment loaders (ISA 16-bit layouts) ----------------
DEV v16bf load_a_bf16(const unsigned short* __restrict__ arow, int kbase, int g) {
    v8bf lo = *(const v8bf*)(arow + kbase + 8*g);
    v8bf hi = *(const v8bf*)(arow + kbase + 16 + 8*g);
    v16bf a;
#pragma unroll
    for (int i = 0; i < 8; ++i) { a[i] = lo[i]; a[8+i] = hi[i]; }
    return a;
}
DEV v8f wmma_bf16(v16bf a, v16bf b, v8f c) {
    return __builtin_amdgcn_wmma_f32_16x16x32_bf16(false, a, false, b, (short)0, c, false, false);
}

// ---- blocked bf16 GEMM body: 32 rows x 64 cols per wave, out = A @ W^T ----
// A: (rows) x Klen bf16 row-major; W: (cols) x Klen bf16 row-major.
template <int KLEN>
DEV void gemm_tile_32x64(const unsigned short* __restrict__ arow0,
                         const unsigned short* __restrict__ brow0,
                         int g, v8f acc[2][4]) {
    const unsigned short* arow1 = arow0 + (size_t)16 * KLEN;
    for (int k = 0; k < KLEN; k += 32) {
        v16bf a0 = load_a_bf16(arow0, k, g);
        v16bf a1 = load_a_bf16(arow1, k, g);
        v16bf b0 = load_a_bf16(brow0 + (size_t)0  * KLEN, k, g);
        v16bf b1 = load_a_bf16(brow0 + (size_t)16 * KLEN, k, g);
        v16bf b2 = load_a_bf16(brow0 + (size_t)32 * KLEN, k, g);
        v16bf b3 = load_a_bf16(brow0 + (size_t)48 * KLEN, k, g);
        acc[0][0] = wmma_bf16(a0, b0, acc[0][0]);
        acc[1][0] = wmma_bf16(a1, b0, acc[1][0]);
        acc[0][1] = wmma_bf16(a0, b1, acc[0][1]);
        acc[1][1] = wmma_bf16(a1, b1, acc[1][1]);
        acc[0][2] = wmma_bf16(a0, b2, acc[0][2]);
        acc[1][2] = wmma_bf16(a1, b2, acc[1][2]);
        acc[0][3] = wmma_bf16(a0, b3, acc[0][3]);
        acc[1][3] = wmma_bf16(a1, b3, acc[1][3]);
        // prefetch 128 bf16 (256B) ahead (global_prefetch_b8 on gfx1250)
        if ((k & 63) == 0 && (k + 128) < KLEN) {
            __builtin_prefetch(arow0 + k + 128, 0, 0);
            __builtin_prefetch(arow1 + k + 128, 0, 0);
            __builtin_prefetch(brow0 + (size_t)0  * KLEN + k + 128, 0, 0);
            __builtin_prefetch(brow0 + (size_t)16 * KLEN + k + 128, 0, 0);
            __builtin_prefetch(brow0 + (size_t)32 * KLEN + k + 128, 0, 0);
            __builtin_prefetch(brow0 + (size_t)48 * KLEN + k + 128, 0, 0);
        }
    }
}

// ---------------- GEMM1: xz = xg @ in_w^T  (M=256,N=4096,K=1024) ----------------
__global__ void k_gemm_in(const unsigned short* __restrict__ xg,
                          const unsigned short* __restrict__ win,
                          unsigned short* __restrict__ xz) {
    const int lane = threadIdx.x & 31, wv = threadIdx.x >> 5;
    const int ng = blockIdx.x*4 + wv;          // 64-col group: 0..63
    const int mb = blockIdx.y;                 // 32-row block: 0..7
    const int bk = blockIdx.z, kd = bk & 7;
    const int m = lane & 15, g = lane >> 4;
    const unsigned short* arow0 = xg  + ((size_t)bk*L_  + mb*32 + m) * CHS;
    const unsigned short* brow0 = win + ((size_t)kd*NIN + ng*64 + m) * CHS;
    v8f acc[2][4] = {};
    gemm_tile_32x64<CHS>(arow0, brow0, g, acc);
    unsigned short* ob = xz + ((size_t)bk*L_ + mb*32) * NIN + ng*64 + m;
#pragma unroll
    for (int i = 0; i < 2; ++i)
#pragma unroll
        for (int j = 0; j < 4; ++j)
#pragma unroll
            for (int r = 0; r < 8; ++r)
                ob[(size_t)(i*16 + r + 8*g) * NIN + j*16] = f2bf(acc[i][j][r]);
}

// ---------------- depthwise conv (d_conv=4) + SiLU -> u ----------------
__global__ void k_conv(const unsigned short* __restrict__ xz, const float* __restrict__ cw,
                       const float* __restrict__ cb, unsigned short* __restrict__ u) {
    int total = B_ * K8 * L_ * DIN;
    for (int t = blockIdx.x * blockDim.x + threadIdx.x; t < total; t += gridDim.x * blockDim.x) {
        int d = t % DIN; int r = t / DIN; int l = r % L_; int bk = r / L_; int k = bk & 7;
        const float* w = cw + ((size_t)k*DIN + d) * 4;
        float acc = cb[k*DIN + d];
        const unsigned short* base = xz + ((size_t)bk * L_) * NIN + d;
#pragma unroll
        for (int j = 0; j < 4; ++j) {
            int ls = l + j - 3;
            if (ls >= 0) acc += w[j] * bf2f(base[(size_t)ls * NIN]);
        }
        u[((size_t)bk*L_ + l) * DIN + d] = f2bf(siluf_(acc));
    }
}

// ---------------- GEMM2: x_dbl = u @ xp_w^T  (M=256,N=96,K=2048) ----------------
__global__ void k_gemm_xp(const unsigned short* __restrict__ u,
                          const unsigned short* __restrict__ wxp,
                          float* __restrict__ xdbl) {
    const int lane = threadIdx.x;
    const int nt = blockIdx.x, mt = blockIdx.y, bk = blockIdx.z, kd = bk & 7;
    const int m = lane & 15, g = lane >> 4;
    const unsigned short* arow = u   + ((size_t)bk*L_  + mt*16 + m) * DIN;
    const unsigned short* brow = wxp + ((size_t)kd*XDB + nt*16 + m) * DIN;
    v8f acc = {};
    for (int k = 0; k < DIN; k += 32) {
        v16bf a  = load_a_bf16(arow, k, g);
        v16bf bm = load_a_bf16(brow, k, g);
        acc = wmma_bf16(a, bm, acc);
        if ((k & 63) == 0 && (k + 128) < DIN) {
            __builtin_prefetch(arow + k + 128, 0, 0);
            __builtin_prefetch(brow + k + 128, 0, 0);
        }
    }
    float* o = xdbl + ((size_t)bk*L_ + mt*16) * XDB + nt*16 + m;
#pragma unroll
    for (int r = 0; r < 8; ++r) o[(size_t)(r + 8*g) * XDB] = acc[r];
}

// ---------------- GEMM3: dt = softplus(x_dbl[:, :64] @ dtp_w^T + b) ----------------
__global__ void k_gemm_dt(const float* __restrict__ xdbl, const unsigned short* __restrict__ wdt,
                          const float* __restrict__ dtp_b, unsigned short* __restrict__ dt) {
    const int lane = threadIdx.x & 31, wv = threadIdx.x >> 5;
    const int nt = blockIdx.x*4 + wv, mt = blockIdx.y, bk = blockIdx.z, kd = bk & 7;
    const int m = lane & 15, g = lane >> 4;
    const float* ar = xdbl + ((size_t)bk*L_ + mt*16 + m) * XDB;
    const unsigned short* brow = wdt + ((size_t)kd*DIN + nt*16 + m) * 64;
    v8f acc = {};
#pragma unroll
    for (int k = 0; k < 64; k += 32) {
        float4 f0 = *(const float4*)(ar + k + 8*g);
        float4 f1 = *(const float4*)(ar + k + 8*g + 4);
        float4 f2 = *(const float4*)(ar + k + 16 + 8*g);
        float4 f3 = *(const float4*)(ar + k + 16 + 8*g + 4);
        v16us t;
        t[0]=f2bf(f0.x);  t[1]=f2bf(f0.y);  t[2]=f2bf(f0.z);  t[3]=f2bf(f0.w);
        t[4]=f2bf(f1.x);  t[5]=f2bf(f1.y);  t[6]=f2bf(f1.z);  t[7]=f2bf(f1.w);
        t[8]=f2bf(f2.x);  t[9]=f2bf(f2.y);  t[10]=f2bf(f2.z); t[11]=f2bf(f2.w);
        t[12]=f2bf(f3.x); t[13]=f2bf(f3.y); t[14]=f2bf(f3.z); t[15]=f2bf(f3.w);
        v16bf a  = __builtin_bit_cast(v16bf, t);
        v16bf bm = load_a_bf16(brow, k, g);
        acc = wmma_bf16(a, bm, acc);
    }
    int n = nt*16 + m;
    float bias = dtp_b[kd*DIN + n];
    unsigned short* o = dt + ((size_t)bk*L_ + mt*16) * DIN + n;
#pragma unroll
    for (int r = 0; r < 8; ++r) {
        float v = acc[r] + bias;
        v = (v > 20.f) ? v : log1pf(__expf(v));
        o[(size_t)(r + 8*g) * DIN] = f2bf(v);
    }
}

// ---------------- selective scan over L + D-skip + z-gate -> bf16 yg ----------------
__global__ void k_sscan(const unsigned short* __restrict__ u, const unsigned short* __restrict__ dt,
                        const float* __restrict__ xdbl, const unsigned short* __restrict__ xz,
                        const float* __restrict__ A_log, const float* __restrict__ Dp,
                        unsigned short* __restrict__ yg) {
    int d = blockIdx.x * 256 + threadIdx.x;
    int k = blockIdx.y, b = blockIdx.z, bk = b*8 + k;
    float A[16], h[16];
#pragma unroll
    for (int s = 0; s < 16; ++s) {
        A[s] = -__expf(A_log[((size_t)k*DIN + d)*16 + s]);
        h[s] = 0.f;
    }
    float Dv = Dp[k*DIN + d];
    __shared__ float sB[16], sC[16];
    const size_t rowbase = (size_t)bk * L_;
    for (int l = 0; l < L_; ++l) {
        if (threadIdx.x < 32) {
            float v = xdbl[(rowbase + l)*XDB + 64 + threadIdx.x];
            if (threadIdx.x < 16) sB[threadIdx.x] = v; else sC[threadIdx.x - 16] = v;
        }
        __syncthreads();
        float dtv = bf2f(dt[(rowbase + l)*DIN + d]);
        float uv  = bf2f(u [(rowbase + l)*DIN + d]);
        float du  = dtv * uv;
        float y = 0.f;
#pragma unroll
        for (int s = 0; s < 16; ++s) {
            h[s] = __expf(dtv * A[s]) * h[s] + du * sB[s];
            y += h[s] * sC[s];
        }
        float zv = bf2f(xz[(rowbase + l)*NIN + DIN + d]);
        yg[(rowbase + l)*DIN + d] = f2bf((y + uv*Dv) * siluf_(zv));
        __syncthreads();
    }
}

// ---------------- GEMM4: ys = yg @ out_w^T  (M=256,N=1024,K=2048) ----------------
__global__ void k_gemm_out(const unsigned short* __restrict__ yg,
                           const unsigned short* __restrict__ wout,
                           float* __restrict__ ys) {
    const int lane = threadIdx.x & 31, wv = threadIdx.x >> 5;
    const int ng = blockIdx.x*4 + wv;          // 64-col group: 0..15
    const int mb = blockIdx.y;                 // 32-row block: 0..7
    const int bk = blockIdx.z, kd = bk & 7;
    const int m = lane & 15, g = lane >> 4;
    const unsigned short* arow0 = yg   + ((size_t)bk*L_  + mb*32 + m) * DIN;
    const unsigned short* brow0 = wout + ((size_t)kd*CHS + ng*64 + m) * DIN;
    v8f acc[2][4] = {};
    gemm_tile_32x64<DIN>(arow0, brow0, g, acc);
    float* ob = ys + ((size_t)bk*L_ + mb*32) * CHS + ng*64 + m;
#pragma unroll
    for (int i = 0; i < 2; ++i)
#pragma unroll
        for (int j = 0; j < 4; ++j)
#pragma unroll
            for (int r = 0; r < 8; ++r)
                ob[(size_t)(i*16 + r + 8*g) * CHS + j*16] = acc[i][j][r];
}

// ---------------- cross-merge -> output (B,CHS,H,W) ----------------
__global__ void k_merge(const float* __restrict__ ys, const float* __restrict__ gw,
                        float* __restrict__ out) {
    int total = B_ * CHS * 256;
    for (int t = blockIdx.x * blockDim.x + threadIdx.x; t < total; t += gridDim.x * blockDim.x) {
        int pos = t % 256, bd = t / 256;
        int d = bd % CHS, b = bd / CHS;
        int h = pos >> 4, w = pos & 15;
        const float* Y = ys + (size_t)b * 8 * L_ * CHS;
        const float* G = gw + b*8;
        int l0 = pos;
        int lT = w*16 + h;
        int ld = (((w - h) & 15) * 16) + h;
        int la = (((w + h) & 15) * 16) + h;
        float acc;
        acc  = G[0]*Y[((size_t)0*L_ +        l0)*CHS + d] + G[2]*Y[((size_t)2*L_ + (255-l0))*CHS + d];
        acc += G[1]*Y[((size_t)1*L_ +        lT)*CHS + d] + G[3]*Y[((size_t)3*L_ + (255-lT))*CHS + d];
        acc += G[4]*Y[((size_t)4*L_ +        ld)*CHS + d] + G[6]*Y[((size_t)6*L_ + (255-ld))*CHS + d];
        acc += G[5]*Y[((size_t)5*L_ +        la)*CHS + d] + G[7]*Y[((size_t)7*L_ + (255-la))*CHS + d];
        out[((size_t)b * CHS + d) * 256 + pos] = acc;
    }
}

// ---------------- host launcher ----------------
extern "C" void kernel_launch(void* const* d_in, const int* in_sizes, int n_in,
                              void* d_out, int out_size, void* d_ws, size_t ws_size,
                              hipStream_t stream) {
    (void)in_sizes; (void)n_in; (void)out_size; (void)ws_size;
    const float* x      = (const float*)d_in[0];
    const float* pw     = (const float*)d_in[1];
    const float* bn_g   = (const float*)d_in[2];
    const float* bn_b   = (const float*)d_in[3];
    const float* gate_w = (const float*)d_in[4];
    const float* ln_g   = (const float*)d_in[5];
    const float* ln_b   = (const float*)d_in[6];
    const float* fd_sc  = (const float*)d_in[7];
    const float* in_w   = (const float*)d_in[8];
    const float* cw     = (const float*)d_in[9];
    const float* cb     = (const float*)d_in[10];
    const float* xp_w   = (const float*)d_in[11];
    const float* dtp_w  = (const float*)d_in[12];
    const float* dtp_b  = (const float*)d_in[13];
    const float* A_log  = (const float*)d_in[14];
    const float* D_p    = (const float*)d_in[15];
    const float* out_w  = (const float*)d_in[16];

    char* w = (char*)d_ws;
    float* xc            = (float*)(w + OFF_XC);
    float* gw            = (float*)(w + OFF_GW);
    float* pool          = (float*)(w + OFF_POOL);
    float* ybr           = (float*)(w + OFF_YBR);
    unsigned short* xg   = (unsigned short*)(w + OFF_XG);
    unsigned short* win  = (unsigned short*)(w + OFF_WIN);
    unsigned short* wxp  = (unsigned short*)(w + OFF_WXP);
    unsigned short* wdt  = (unsigned short*)(w + OFF_WDT);
    unsigned short* wout = (unsigned short*)(w + OFF_WOUT);
    unsigned short* xz   = (unsigned short*)(w + OFF_XZ);
    unsigned short* u    = (unsigned short*)(w + OFF_U);
    float* xdbl          = (float*)(w + OFF_XDBL);
    unsigned short* dt   = (unsigned short*)(w + OFF_DT);
    unsigned short* yg   = (unsigned short*)(w + OFF_YG);
    float* ys            = (float*)(w + OFF_YS);
    float* out           = (float*)d_out;

    // weight conversion to bf16
    k_cvt<<<4096, 256, 0, stream>>>(in_w,  win,  K8*NIN*CHS);
    k_cvt<<<1024, 256, 0, stream>>>(xp_w,  wxp,  K8*XDB*DIN);
    k_cvt<<<1024, 256, 0, stream>>>(dtp_w, wdt,  K8*DIN*64);
    k_cvt<<<4096, 256, 0, stream>>>(out_w, wout, K8*CHS*DIN);

    // pooling pyramid + CBR + concat -> xc
    k_pool<<<4400, 256, 0, stream>>>(x, pool);
    k_cbr<<<1100, 256, 0, stream>>>(pool, pw, bn_g, bn_b, ybr);
    k_p0<<<B_*DIM, 256, 0, stream>>>(x, pw, bn_g, bn_b, xc);
    k_assemble<<<7168, 256, 0, stream>>>(x, ybr, xc);

    // expert gating
    k_gate<<<B_, 256, 0, stream>>>(xc, gate_w, gw);

    // cross-scan + fd gate -> bf16 activations
    k_scanfd<<<dim3(L_, K8, B_), 256, 0, stream>>>(xc, ln_g, ln_b, fd_sc, xg);

    // in_proj (WMMA bf16, 32x64 wave tiles)
    k_gemm_in<<<dim3(16, 8, B_*K8), 128, 0, stream>>>(xg, win, xz);

    // depthwise conv + SiLU
    k_conv<<<32768, 256, 0, stream>>>(xz, cw, cb, u);

    // x_proj (WMMA bf16)
    k_gemm_xp<<<dim3(XDB/16, L_/16, B_*K8), 32, 0, stream>>>(u, wxp, xdbl);

    // dt_proj + softplus (WMMA bf16)
    k_gemm_dt<<<dim3(DIN/64, L_/16, B_*K8), 128, 0, stream>>>(xdbl, wdt, dtp_b, dt);

    // selective state-space scan + gate
    k_sscan<<<dim3(DIN/256, K8, B_), 256, 0, stream>>>(u, dt, xdbl, xz, A_log, D_p, yg);

    // out_proj (WMMA bf16, 32x64 wave tiles)
    k_gemm_out<<<dim3(4, 8, B_*K8), 128, 0, stream>>>(yg, wout, ys);

    // cross-merge with gate weights -> output
    k_merge<<<8192, 256, 0, stream>>>(ys, gw, out);
}
